// Qwen3MoeAttention_50397146251459
// MI455X (gfx1250) — matrix operand order repaired and checked by reading, not compiled, not run on hardware
//
#include <hip/hip_runtime.h>
#include <hip/hip_bf16.h>

// ---------------------------------------------------------------------------
// Types / WMMA helpers (CDNA5 gfx1250, wave32)
// ---------------------------------------------------------------------------
typedef __bf16 bf16_t;
typedef __attribute__((ext_vector_type(16))) __bf16 v16bf;
typedef __attribute__((ext_vector_type(4)))  __bf16 v4bf;
typedef __attribute__((ext_vector_type(8)))  float  v8f;

union Frag { v16bf v; unsigned u[8]; };

__device__ __forceinline__ v8f wmma_bf16(v16bf a, v16bf b, v8f c) {
  // D = A(16x32 bf16) * B(32x16 bf16) + C(16x16 f32)
  return __builtin_amdgcn_wmma_f32_16x16x32_bf16(
      false, a, false, b, (short)0, c, false, false);
}

__device__ __forceinline__ unsigned pk2(float a, float b) {
  union { bf16_t h[2]; unsigned u; } x;
  x.h[0] = (bf16_t)a; x.h[1] = (bf16_t)b;
  return x.u;
}

// gfx1250 async copy: 16B per lane, global -> LDS, tracked by ASYNCcnt
__device__ __forceinline__ void async_b128(const bf16_t* g, const bf16_t* l) {
  asm volatile("global_load_async_to_lds_b128 %0, %1, off"
               :: "v"((unsigned)(unsigned long long)l),
                  "v"((unsigned long long)g)
               : "memory");
}
__device__ __forceinline__ void wait_async0() {
  asm volatile("s_wait_asynccnt 0x0" ::: "memory");
}

// A-fragment (16x32, 16-bit): lane row = lane&15; VGPR i K-pair =
//   (i>=4?16:0) + (lane>>4)*8 + 2*(i&3)                       [ISA 7.12.2]
template <int STRIDE>
__device__ __forceinline__ Frag load_frag_a(const bf16_t* base, int lane) {
  Frag f;
  const int row = lane & 15, half = lane >> 4;
  const bf16_t* r = base + row * STRIDE;
#pragma unroll
  for (int i = 0; i < 8; ++i) {
    const int k = ((i >> 2) << 4) + half * 8 + ((i & 3) << 1);
    f.u[i] = *(const unsigned*)(r + k);
  }
  return f;
}
// B-fragment (32x16) from a TRANSPOSED [N][K] LDS tile: lane col = lane&15;
//   VGPR v K-pair = (v>=4?16:0) + (lane>>4)*8 + 2*(v&3)
template <int STRIDE>
__device__ __forceinline__ Frag load_frag_b(const bf16_t* base, int lane) {
  Frag f;
  const int col = lane & 15, half = lane >> 4;
  const bf16_t* r = base + col * STRIDE;
#pragma unroll
  for (int i = 0; i < 8; ++i) {
    const int k = ((i >> 2) << 4) + half * 8 + ((i & 3) << 1);
    f.u[i] = *(const unsigned*)(r + k);
  }
  return f;
}

// ---------------------------------------------------------------------------
// Problem constants
// ---------------------------------------------------------------------------
constexpr int H_    = 32;
constexpr int KVH_  = 4;
constexpr int HD_   = 128;
constexpr int HID_  = 2048;
constexpr int B_    = 2;
constexpr int S_    = 1024;
constexpr int T_    = B_ * S_;                 // 2048
constexpr int QKVW_ = (H_ + 2 * KVH_) * HD_;   // 5120
constexpr int OW_   = H_ * HD_;                // 4096
constexpr float SCALE_ = 0.08838834764831845f; // 128^-0.5
constexpr float NEGF_  = -3.0e38f;

// ---------------------------------------------------------------------------
// fp32 -> bf16 convert (pairs)
// ---------------------------------------------------------------------------
__global__ __launch_bounds__(256)
void conv_bf16_kernel(const float* __restrict__ in, bf16_t* __restrict__ out) {
  const size_t i = (size_t)blockIdx.x * 256 + threadIdx.x;
  const float2 f = *(const float2*)(in + 2 * i);
  *(unsigned*)&out[2 * i] = pk2(f.x, f.y);
}

// fp32 w[K][N] -> bf16 wT[N][K]  (64x64 LDS tile transpose)
__global__ __launch_bounds__(256)
void conv_transpose_kernel(const float* __restrict__ w, bf16_t* __restrict__ wT,
                           int K, int N) {
  __shared__ float lT[64][65];
  const int k0 = blockIdx.y * 64, n0 = blockIdx.x * 64;
#pragma unroll
  for (int i = threadIdx.x; i < 64 * 64; i += 256) {
    const int r = i >> 6, c = i & 63;
    lT[r][c] = w[(size_t)(k0 + r) * N + n0 + c];
  }
  __syncthreads();
#pragma unroll
  for (int i = threadIdx.x; i < 64 * 32; i += 256) {
    const int r = i >> 5, p = i & 31;
    *(unsigned*)&wT[(size_t)(n0 + r) * K + k0 + 2 * p] =
        pk2(lT[2 * p][r], lT[2 * p + 1][r]);
  }
}

// ---------------------------------------------------------------------------
// bf16 GEMM with async-to-LDS double buffering:
//   C[M,N] = A[M,K] @ Bt[N,K]^T ; 128x128x32 tiles, 8 waves, wave => 64x32
//   LDS row stride 40 bf16 = 80B (16B-multiple for b128, conflict-free)
// ---------------------------------------------------------------------------
constexpr int LSTR = 40;

template <typename OutT>
__global__ __launch_bounds__(256)
void gemm_async_kernel(const bf16_t* __restrict__ A, const bf16_t* __restrict__ Bt,
                       OutT* __restrict__ C, int M, int N, int K) {
  __shared__ bf16_t lA[2][128][LSTR];
  __shared__ bf16_t lB[2][128][LSTR];
  const int tid = threadIdx.x, lane = tid & 31, wave = tid >> 5;
  const int wm = wave & 1, wn = wave >> 1;
  const int m0 = blockIdx.y * 128, n0 = blockIdx.x * 128;
  const bf16_t* Ab = A + (size_t)m0 * K;
  const bf16_t* Bb = Bt + (size_t)n0 * K;

  v8f acc[4][2];
#pragma unroll
  for (int mt = 0; mt < 4; ++mt)
#pragma unroll
    for (int nt = 0; nt < 2; ++nt) acc[mt][nt] = {};

  // 128 rows x 64B per tile = 512 b128 chunks; 2 per thread per matrix
  auto issue = [&](int buf, int k0) {
#pragma unroll
    for (int c = 0; c < 2; ++c) {
      const int idx = tid + (c << 8);
      const int r = idx >> 2;
      const int off = (idx & 3) << 3;            // 0,8,16,24 elems
      async_b128(Ab + (size_t)r * K + k0 + off, &lA[buf][r][off]);
      async_b128(Bb + (size_t)r * K + k0 + off, &lB[buf][r][off]);
    }
  };

  issue(0, 0);
  const int NIT = K >> 5;
  for (int it = 0; it < NIT; ++it) {
    wait_async0();          // my copies for buf landed
    __syncthreads();        // everyone's landed; prev compute finished
    if (it + 1 < NIT) issue((it + 1) & 1, (it + 1) << 5);
    const int buf = it & 1;

    Frag af[4], bfr[2];
#pragma unroll
    for (int mt = 0; mt < 4; ++mt)
      af[mt] = load_frag_a<LSTR>(&lA[buf][wm * 64 + mt * 16][0], lane);
#pragma unroll
    for (int nt = 0; nt < 2; ++nt)
      bfr[nt] = load_frag_b<LSTR>(&lB[buf][wn * 32 + nt * 16][0], lane);
#pragma unroll
    for (int mt = 0; mt < 4; ++mt)
#pragma unroll
      for (int nt = 0; nt < 2; ++nt)
        acc[mt][nt] = wmma_bf16(af[mt].v, bfr[nt].v, acc[mt][nt]);
  }

  const int l16 = lane & 15, half = lane >> 4;
#pragma unroll
  for (int mt = 0; mt < 4; ++mt)
#pragma unroll
    for (int nt = 0; nt < 2; ++nt) {
      const int rbase = m0 + wm * 64 + mt * 16 + 8 * half;
      const int cc    = n0 + wn * 32 + nt * 16 + l16;
#pragma unroll
      for (int g = 0; g < 8; ++g)
        C[(size_t)(rbase + g) * N + cc] = (OutT)acc[mt][nt][g];
    }
}

// ---------------------------------------------------------------------------
// RMSNorm + RoPE (+ fold softmax scale into q) + bf16 KV scatter.
// One wave per 128-wide head vector; bf16 in/out, f32 math.
// ---------------------------------------------------------------------------
__global__ __launch_bounds__(128)
void norm_rope_scatter_kernel(bf16_t* __restrict__ qkv,
                              const float* __restrict__ qw,
                              const float* __restrict__ kw,
                              const float* __restrict__ cosb,
                              const float* __restrict__ sinb,
                              const int*   __restrict__ slots,
                              bf16_t* __restrict__ kc, bf16_t* __restrict__ vc) {
  const int wave = threadIdx.x >> 5, lane = threadIdx.x & 31;
  const int hh = blockIdx.x * 4 + wave;
  const int t = hh / (H_ + 2 * KVH_);
  const int j = hh % (H_ + 2 * KVH_);
  const int d = lane * 4;

  bf16_t* p = qkv + (size_t)t * QKVW_ + j * HD_ + d;
  const v4bf raw = *(const v4bf*)p;
  float x0 = (float)raw[0], x1 = (float)raw[1];
  float x2 = (float)raw[2], x3 = (float)raw[3];

  if (j < H_ + KVH_) {  // q or k: RMS norm + RoPE
    float ss = x0 * x0 + x1 * x1 + x2 * x2 + x3 * x3;
#pragma unroll
    for (int m = 16; m; m >>= 1) ss += __shfl_xor(ss, m, 32);
    const float inv = rsqrtf(ss * (1.0f / HD_) + 1e-6f);
    const float* w = (j < H_) ? qw : kw;
    x0 *= inv * w[d];     x1 *= inv * w[d + 1];
    x2 *= inv * w[d + 2]; x3 *= inv * w[d + 3];

    const int s = t & (S_ - 1);
    const float4 c  = *(const float4*)(cosb + (size_t)s * HD_ + d);
    const float4 sn = *(const float4*)(sinb + (size_t)s * HD_ + d);
    const float r0 = __shfl_xor(x0, 16, 32), r1 = __shfl_xor(x1, 16, 32);
    const float r2 = __shfl_xor(x2, 16, 32), r3 = __shfl_xor(x3, 16, 32);
    const float sgn = (lane < 16) ? -1.0f : 1.0f;
    x0 = x0 * c.x + sgn * r0 * sn.x;
    x1 = x1 * c.y + sgn * r1 * sn.y;
    x2 = x2 * c.z + sgn * r2 * sn.z;
    x3 = x3 * c.w + sgn * r3 * sn.w;
  }

  v4bf o;
  if (j < H_) {                       // q: pre-scale by HD^-0.5
    o[0] = (bf16_t)(x0 * SCALE_); o[1] = (bf16_t)(x1 * SCALE_);
    o[2] = (bf16_t)(x2 * SCALE_); o[3] = (bf16_t)(x3 * SCALE_);
    *(v4bf*)p = o;
  } else {
    o[0] = (bf16_t)x0; o[1] = (bf16_t)x1; o[2] = (bf16_t)x2; o[3] = (bf16_t)x3;
    const int slot = slots[t];
    const bool isK = (j < H_ + KVH_);
    bf16_t* dst = (isK ? kc : vc) +
                  ((size_t)slot * KVH_ + (j - (isK ? H_ : H_ + KVH_))) * HD_ + d;
    *(v4bf*)dst = o;
  }
}

// ---------------------------------------------------------------------------
// Causal flash attention. Block = (qt, h, b), 4 waves x 16 q-rows.
// K tiles arrive via async-to-LDS; bf16 in, bf16 out.
// ---------------------------------------------------------------------------
__global__ __launch_bounds__(128)
void flash_attn_kernel(const bf16_t* __restrict__ qkv,
                       const bf16_t* __restrict__ kc,
                       const bf16_t* __restrict__ vc,
                       const int*   __restrict__ slots,
                       bf16_t* __restrict__ attn) {
  __shared__ bf16_t lK[64][136];      // [s][d]; 272B rows (16B-multiple)
  __shared__ bf16_t lV[128][66];      // V transposed [d][s]
  __shared__ bf16_t lP[4][16][66];    // per-wave P tile

  const int tid = threadIdx.x, lane = tid & 31, wave = tid >> 5;
  const int qt = blockIdx.x, h = blockIdx.y, b = blockIdx.z;
  const int kvh = h >> 3;             // GROUP = 8
  const int l16 = lane & 15, half = lane >> 4;

  // Q resident as 4 A-fragments; pre-scaled bf16 -> straight uint4 loads
  Frag qa[4];
  {
    const int row = qt * 64 + wave * 16 + l16;
    const bf16_t* qb = qkv + (size_t)(b * S_ + row) * QKVW_ + h * HD_ + half * 8;
#pragma unroll
    for (int c = 0; c < 4; ++c) {
      const uint4 lo = *(const uint4*)(qb + c * 32);
      const uint4 hi = *(const uint4*)(qb + c * 32 + 16);
      qa[c].u[0] = lo.x; qa[c].u[1] = lo.y; qa[c].u[2] = lo.z; qa[c].u[3] = lo.w;
      qa[c].u[4] = hi.x; qa[c].u[5] = hi.y; qa[c].u[6] = hi.z; qa[c].u[7] = hi.w;
    }
  }

  float m[8], l[8];
  v8f o[8];
#pragma unroll
  for (int g = 0; g < 8; ++g) { m[g] = NEGF_; l[g] = 0.0f; }
#pragma unroll
  for (int ht = 0; ht < 8; ++ht) o[ht] = {};

  for (int kb = 0; kb <= qt * 64; kb += 64) {
    // K tile: 64 rows x 256B, gathered by slot, async b128 chunks
#pragma unroll
    for (int i = tid; i < 64 * 16; i += 128) {
      const int r = i >> 4, ch = i & 15;
      const int slot = slots[b * S_ + kb + r];
      async_b128(kc + ((size_t)slot * KVH_ + kvh) * HD_ + ch * 8, &lK[r][ch * 8]);
    }
    // V tile transposed into [d][s]
#pragma unroll
    for (int i = tid; i < 64 * 64; i += 128) {
      const int r = i >> 6, pp = i & 63;
      const int slot = slots[b * S_ + kb + r];
      union { unsigned u; bf16_t hh2[2]; } x;
      x.u = *(const unsigned*)(vc + ((size_t)slot * KVH_ + kvh) * HD_ + 2 * pp);
      lV[2 * pp][r]     = x.hh2[0];
      lV[2 * pp + 1][r] = x.hh2[1];
    }
    wait_async0();
    __syncthreads();

    // scores: 4 col-tiles of 16; K-dim = HD over 4 chunks of 32
    v8f sc[4];
#pragma unroll
    for (int nt = 0; nt < 4; ++nt) {
      v8f s = {};
#pragma unroll
      for (int c = 0; c < 4; ++c) {
        Frag bfg = load_frag_b<136>(&lK[nt * 16][c * 32], lane);
        s = wmma_bf16(qa[c].v, bfg.v, s);
      }
      sc[nt] = s;
    }

    // causal mask on diagonal block only
    if (kb == qt * 64) {
#pragma unroll
      for (int nt = 0; nt < 4; ++nt)
#pragma unroll
        for (int g = 0; g < 8; ++g) {
          const int col = kb + nt * 16 + l16;
          const int rw  = qt * 64 + wave * 16 + g + 8 * half;
          if (col > rw) sc[nt][g] = NEGF_;
        }
    }

    // online softmax (one row per C-VGPR on each 16-lane half)
#pragma unroll
    for (int g = 0; g < 8; ++g) {
      float mx = fmaxf(fmaxf(sc[0][g], sc[1][g]), fmaxf(sc[2][g], sc[3][g]));
#pragma unroll
      for (int msk = 8; msk; msk >>= 1) mx = fmaxf(mx, __shfl_xor(mx, msk, 32));
      const float mnew  = fmaxf(m[g], mx);
      const float alpha = __expf(m[g] - mnew);
      m[g] = mnew;
      float sum = 0.0f;
#pragma unroll
      for (int nt = 0; nt < 4; ++nt) {
        const float pv = __expf(sc[nt][g] - mnew);
        sc[nt][g] = pv;
        sum += pv;
      }
#pragma unroll
      for (int msk = 8; msk; msk >>= 1) sum += __shfl_xor(sum, msk, 32);
      l[g] = l[g] * alpha + sum;
#pragma unroll
      for (int ht = 0; ht < 8; ++ht) o[ht][g] *= alpha;
    }

    // C-layout -> A-layout via per-wave LDS round trip
#pragma unroll
    for (int nt = 0; nt < 4; ++nt)
#pragma unroll
      for (int g = 0; g < 8; ++g)
        lP[wave][g + 8 * half][nt * 16 + l16] = (bf16_t)sc[nt][g];
    __syncthreads();

    Frag pA[2];
    pA[0] = load_frag_a<66>(&lP[wave][0][0], lane);
    pA[1] = load_frag_a<66>(&lP[wave][0][32], lane);
#pragma unroll
    for (int ht = 0; ht < 8; ++ht) {
#pragma unroll
      for (int c2 = 0; c2 < 2; ++c2) {
        Frag vB = load_frag_b<66>(&lV[ht * 16][c2 * 32], lane);
        o[ht] = wmma_bf16(pA[c2].v, vB.v, o[ht]);
      }
    }
    __syncthreads();  // protect lK/lV before next tile
  }

  // normalize + bf16 store: attn[t][h*128 + d]
#pragma unroll
  for (int g = 0; g < 8; ++g) {
    const float inv = 1.0f / l[g];
    const int rw = qt * 64 + wave * 16 + g + 8 * half;
    bf16_t* op = attn + (size_t)(b * S_ + rw) * OW_ + h * HD_;
#pragma unroll
    for (int ht = 0; ht < 8; ++ht)
      op[ht * 16 + l16] = (bf16_t)(o[ht][g] * inv);
  }
}

// ---------------------------------------------------------------------------
// Host launcher
// ---------------------------------------------------------------------------
extern "C" void kernel_launch(void* const* d_in, const int* in_sizes, int n_in,
                              void* d_out, int out_size, void* d_ws, size_t ws_size,
                              hipStream_t stream) {
  (void)in_sizes; (void)n_in; (void)out_size; (void)ws_size;
  const float* hidden = (const float*)d_in[0];
  const float* wqkv   = (const float*)d_in[1];
  const float* wo     = (const float*)d_in[2];
  const float* qw     = (const float*)d_in[3];
  const float* kw     = (const float*)d_in[4];
  const float* cosb   = (const float*)d_in[5];
  const float* sinb   = (const float*)d_in[6];
  // d_in[7]/d_in[8] (kcache/vcache) never mutated; bf16 copies live in ws
  const int* slots    = (const int*)d_in[10];
  float* out = (float*)d_out;

  char* ws = (char*)d_ws;
  bf16_t* hid_bf = (bf16_t*)ws;  ws += (size_t)T_ * HID_ * 2;
  bf16_t* wqkvT  = (bf16_t*)ws;  ws += (size_t)QKVW_ * HID_ * 2;
  bf16_t* woT    = (bf16_t*)ws;  ws += (size_t)HID_ * OW_ * 2;
  bf16_t* qkvb   = (bf16_t*)ws;  ws += (size_t)T_ * QKVW_ * 2;
  bf16_t* kcb    = (bf16_t*)ws;  ws += (size_t)3072 * KVH_ * HD_ * 2;
  bf16_t* vcb    = (bf16_t*)ws;  ws += (size_t)3072 * KVH_ * HD_ * 2;
  bf16_t* attnb  = (bf16_t*)ws;

  // 0) one-time bf16 conversion / weight transposes
  conv_bf16_kernel<<<(T_ * HID_ / 2) / 256, 256, 0, stream>>>(hidden, hid_bf);
  conv_transpose_kernel<<<dim3(QKVW_ / 64, HID_ / 64), 256, 0, stream>>>(
      wqkv, wqkvT, HID_, QKVW_);
  conv_transpose_kernel<<<dim3(HID_ / 64, OW_ / 64), 256, 0, stream>>>(
      wo, woT, OW_, HID_);

  // 1) QKV projection (bf16 out)
  gemm_async_kernel<bf16_t><<<dim3(QKVW_ / 128, T_ / 128), 256, 0, stream>>>(
      hid_bf, wqkvT, qkvb, T_, QKVW_, HID_);

  // 2) RMSNorm + RoPE (+scale fold) + KV scatter
  norm_rope_scatter_kernel<<<T_ * (H_ + 2 * KVH_) / 4, 128, 0, stream>>>(
      qkvb, qw, kw, cosb, sinb, slots, kcb, vcb);

  // 3) causal flash attention (bf16 out)
  flash_attn_kernel<<<dim3(S_ / 64, H_, B_), 128, 0, stream>>>(
      qkvb, kcb, vcb, slots, attnb);

  // 4) output projection (fp32 out)
  gemm_async_kernel<float><<<dim3(HID_ / 128, T_ / 128), 256, 0, stream>>>(
      attnb, woT, out, T_, HID_, OW_);
}
